// EquiformerV2Backbone_78589311582941
// MI455X (gfx1250) — matrix operand order repaired and checked by reading, not compiled, not run on hardware
//
#include <hip/hip_runtime.h>

typedef unsigned short u16;
typedef __bf16 bf16_t;
typedef bf16_t v16bf __attribute__((ext_vector_type(16)));
typedef float  v8f   __attribute__((ext_vector_type(8)));

#define NN   2500
#define EE   50000
#define CC   128
#define L2Q  25
#define HH   8
#define AA   64
#define VVD  16
#define FFD  128
#define NLAY 2
#define NB_  600
#define KE   864     // 856 padded to multiple of 32
#define EPB  16
#define NPB  4

__device__ __forceinline__ u16 f2bf(float x){
  unsigned u = __float_as_uint(x);
  unsigned r = u + 0x7fffu + ((u >> 16) & 1u);
  return (u16)(r >> 16);
}
__device__ __forceinline__ float bf2f(unsigned h){ return __uint_as_float(h << 16); }
// fast silu: x * v_rcp_f32(1+exp(-x)) -- avoids the IEEE div expansion
__device__ __forceinline__ float siluf(float x){
  return x * __builtin_amdgcn_rcpf(1.0f + __expf(-x));
}

union FragU { unsigned u[8]; v16bf v; };

__device__ __forceinline__ v8f wmma_bf16(v16bf a, v16bf b, v8f c){
  return __builtin_amdgcn_wmma_f32_16x16x32_bf16(false, a, false, b, (short)0, c, false, false);
}

// A fragment: 16x32 tile, rows [m0,m0+16), K window [kofs,kofs+32), row-major bf16 source.
// lane<16: m=m0+lane, K pairs {0,2,4,6}+{16,18,20,22}; lane>=16: +8 on K base.
__device__ __forceinline__ v16bf frag_a(const u16* base, int m0, int stride, int kofs, int lane){
  const u16* r = base + (size_t)(m0 + (lane & 15)) * stride + kofs + ((lane & 16) >> 1);
  FragU f;
#pragma unroll
  for (int i = 0; i < 4; ++i) f.u[i]   = *(const unsigned*)(r + 2*i);
#pragma unroll
  for (int i = 0; i < 4; ++i) f.u[4+i] = *(const unsigned*)(r + 16 + 2*i);
  return f.v;
}
// B fragment: 32x16 tile from TRANSPOSED [N][K] storage (same K order as frag_a: {0..7}+{16..23}, +8 for lane>=16).
__device__ __forceinline__ v16bf frag_b(const u16* baseT, int n0, int stride, int kofs, int lane){
  const u16* r = baseT + (size_t)(n0 + (lane & 15)) * stride + kofs + ((lane & 16) >> 1);
  FragU f;
#pragma unroll
  for (int i = 0; i < 4; ++i) f.u[i]   = *(const unsigned*)(r + 2*i);
#pragma unroll
  for (int i = 0; i < 4; ++i) f.u[4+i] = *(const unsigned*)(r + 16 + 2*i);
  return f.v;
}
// D layout: VGPR v, lane<16 -> m=m0+v, lane>=16 -> m=m0+8+v; n = n0+(lane&15)
__device__ __forceinline__ void store_acc(u16* d, int m0, int n0, int stride, v8f acc, int lane){
  int n = n0 + (lane & 15);
  int mb = m0 + ((lane & 16) >> 1);
#pragma unroll
  for (int v = 0; v < 8; ++v) d[(size_t)(mb + v) * stride + n] = f2bf(acc[v]);
}
__device__ __forceinline__ void store_accT(u16* d, int m0, int n0, int strideT, v8f acc, int lane){
  int n = n0 + (lane & 15);
  int mb = m0 + ((lane & 16) >> 1);
#pragma unroll
  for (int v = 0; v < 8; ++v) d[(size_t)n * strideT + (mb + v)] = f2bf(acc[v]);
}
__device__ __forceinline__ void atomicMaxF(float* a, float v){
  if (v >= 0.0f) atomicMax((int*)a, __float_as_int(v));
  else           atomicMin((unsigned int*)a, __float_as_uint(v));
}

// ---------------- weight convert: W[K,N] f32 -> WT[N][Kpad] bf16 ----------------
__global__ void k_w2bf_t(const float* __restrict__ W, u16* __restrict__ WT,
                         int K, int Nn, int Kpad, int total){
  int idx = blockIdx.x * 256 + threadIdx.x;
  if (idx >= total) return;
  int n = idx / Kpad, k = idx % Kpad;
  WT[idx] = (k < K) ? f2bf(W[(size_t)k * Nn + n]) : (u16)0;
}

// ---------------- node init: x[:,0,:] = atom_emb[z], rest 0 ----------------
__global__ void k_embed(const int* __restrict__ zn, const float* __restrict__ aemb,
                        float* __restrict__ x){
  int idx = blockIdx.x * 256 + threadIdx.x;
  if (idx >= NN * L2Q * CC) return;
  int n = idx / (L2Q * CC);
  int r = idx % (L2Q * CC);
  int l = r >> 7, c = r & 127;
  x[idx] = (l == 0) ? aemb[(size_t)zn[n] * CC + c] : 0.0f;
}

__global__ void k_init_soft(float* __restrict__ nmax, float* __restrict__ nz){
  int idx = blockIdx.x * 256 + threadIdx.x;
  if (idx < NN * HH){ nmax[idx] = -3.0e38f; nz[idx] = 0.0f; }
}

// ---------------- equivariant RMS norm: h = x / rms(x over L2) * gamma ----------------
__global__ void k_rmsnorm(const float* __restrict__ x, const float* __restrict__ gamma,
                          u16* __restrict__ h){
  int idx = blockIdx.x * 256 + threadIdx.x;
  if (idx >= NN * CC) return;
  int n = idx >> 7, c = idx & 127;
  const float* xb = x + (size_t)n * L2Q * CC + c;
  float xs[L2Q]; float s = 0.0f;
#pragma unroll
  for (int l = 0; l < L2Q; ++l){ float v = xb[l * CC]; xs[l] = v; s += v * v; }
  float rinv = rsqrtf(s * (1.0f / L2Q) + 1e-6f);
  float g = gamma[c];
  u16* hb = h + (size_t)n * L2Q * CC + c;
#pragma unroll
  for (int l = 0; l < L2Q; ++l) hb[l * CC] = f2bf(xs[l] * rinv * g);
}

// ---------------- edge features: basis+emb -> silu(silu(.@We1)@We2) ----------------
__global__ void k_edgefeat(const int* __restrict__ ei, const int* __restrict__ zn,
                           const float* __restrict__ pos,
                           const float* __restrict__ semb, const float* __restrict__ demb,
                           const u16* __restrict__ We1T, const u16* __restrict__ We2T,
                           u16* __restrict__ efb){
  extern __shared__ char smem[];
  u16* Atile = (u16*)smem;               // 32*KE
  u16* t1    = Atile + 32 * KE;          // 32*CC
  float* dls = (float*)(t1 + 32 * CC);   // 32
  int* zs    = (int*)(dls + 32);         // 32
  int* zd    = zs + 32;                  // 32
  int t = threadIdx.x, lane = t & 31, w = t >> 5;
  int e0 = blockIdx.x * 32;
  if (t < 32){
    int e = e0 + t; float d = 0.0f; int a = 0, b = 0;
    if (e < EE){
      int s = ei[e], dn = ei[EE + e];
      float dx = pos[s*3+0] - pos[dn*3+0];
      float dy = pos[s*3+1] - pos[dn*3+1];
      float dz = pos[s*3+2] - pos[dn*3+2];
      d = sqrtf(dx*dx + dy*dy + dz*dz);
      a = zn[s]; b = zn[dn];
    }
    dls[t] = d; zs[t] = a; zd[t] = b;
  }
  __syncthreads();
  const float delta = 12.0f / 599.0f;
  const float coeff = -0.5f / ((2.0f * delta) * (2.0f * delta));
  for (int idx = t; idx < 32 * KE; idx += 256){
    int row = idx / KE, k = idx % KE;
    float v = 0.0f;
    if (k < NB_){
      float dd = dls[row] - (float)k * delta;
      v = __expf(coeff * dd * dd);
    } else if (k < NB_ + CC){
      v = semb[(size_t)zs[row] * CC + (k - NB_)];
    } else if (k < NB_ + 2*CC){
      v = demb[(size_t)zd[row] * CC + (k - NB_ - CC)];
    }
    Atile[idx] = f2bf(v);
  }
  __syncthreads();
  v8f c0 = {}, c1 = {};
#pragma unroll 1
  for (int kt = 0; kt < KE / 32; ++kt){
    v16bf b  = frag_b(We1T, w*16, KE, kt*32, lane);
    v16bf a0 = frag_a(Atile, 0,  KE, kt*32, lane);
    v16bf a1 = frag_a(Atile, 16, KE, kt*32, lane);
    c0 = wmma_bf16(a0, b, c0); c1 = wmma_bf16(a1, b, c1);
  }
#pragma unroll
  for (int v = 0; v < 8; ++v){ c0[v] = siluf(c0[v]); c1[v] = siluf(c1[v]); }
  store_acc(t1, 0,  w*16, CC, c0, lane);
  store_acc(t1, 16, w*16, CC, c1, lane);
  __syncthreads();
  v8f d0 = {}, d1 = {};
#pragma unroll
  for (int kt = 0; kt < 4; ++kt){
    v16bf b  = frag_b(We2T, w*16, CC, kt*32, lane);
    v16bf a0 = frag_a(t1, 0,  CC, kt*32, lane);
    v16bf a1 = frag_a(t1, 16, CC, kt*32, lane);
    d0 = wmma_bf16(a0, b, d0); d1 = wmma_bf16(a1, b, d1);
  }
  int n = w*16 + (lane & 15);
  int mb = (lane & 16) >> 1;
#pragma unroll
  for (int v = 0; v < 8; ++v){
    int e  = e0 + mb + v;       if (e  < EE) efb[(size_t)e  * CC + n] = f2bf(siluf(d0[v]));
    int e2 = e0 + 16 + mb + v;  if (e2 < EE) efb[(size_t)e2 * CC + n] = f2bf(siluf(d1[v]));
  }
}

// ---------------- logits: rad = ef@Wrad; m0 = wig_row0 . (h[s]+h[d]); a=silu((m0*rad)@Wa1) ----------------
__global__ void k_logits(const int* __restrict__ ei, const float* __restrict__ wig,
                         const u16* __restrict__ hbuf, const u16* __restrict__ efb,
                         const u16* __restrict__ WradT, const u16* __restrict__ Wa1T,
                         const float* __restrict__ wa2,
                         u16* __restrict__ radb, float* __restrict__ logits,
                         float* __restrict__ nmax){
  extern __shared__ char smem[];
  u16* efA   = (u16*)smem;               // 32*CC, reused as xa
  u16* radS  = efA + 32 * CC;            // 32*CC
  float* lgS = (float*)(radS + 32 * CC); // 32*HH
  int t = threadIdx.x, lane = t & 31, w = t >> 5;
  int e0 = blockIdx.x * 32;
  for (int idx = t; idx < 32 * CC; idx += 256){
    int e = e0 + (idx >> 7);
    efA[idx] = (e < EE) ? efb[(size_t)e * CC + (idx & 127)] : (u16)0;
  }
  lgS[t] = 0.0f;
  __syncthreads();
  { // rad GEMM
    v8f c0 = {}, c1 = {};
#pragma unroll
    for (int kt = 0; kt < 4; ++kt){
      v16bf b  = frag_b(WradT, w*16, CC, kt*32, lane);
      v16bf a0 = frag_a(efA, 0,  CC, kt*32, lane);
      v16bf a1 = frag_a(efA, 16, CC, kt*32, lane);
      c0 = wmma_bf16(a0, b, c0); c1 = wmma_bf16(a1, b, c1);
    }
    store_acc(radS, 0,  w*16, CC, c0, lane);
    store_acc(radS, 16, w*16, CC, c1, lane);
    int n = w*16 + (lane & 15); int mb = (lane & 16) >> 1;
#pragma unroll
    for (int v = 0; v < 8; ++v){
      int e  = e0 + mb + v;      if (e  < EE) radb[(size_t)e  * CC + n] = f2bf(c0[v]);
      int e2 = e0 + 16 + mb + v; if (e2 < EE) radb[(size_t)e2 * CC + n] = f2bf(c1[v]);
    }
  }
  __syncthreads();
  { // xa = (wig row0 . msg) * rad  (overwrite efA); vectorized 16B gathers
    int row = t >> 3, part = t & 7;
    int e = e0 + row;
    float m0[16];
#pragma unroll
    for (int i = 0; i < 16; ++i) m0[i] = 0.0f;
    if (e < EE){
      int s = ei[e], dn = ei[EE + e];
      const u16* hs = hbuf + (size_t)s  * L2Q * CC + part * 16;
      const u16* hd = hbuf + (size_t)dn * L2Q * CC + part * 16;
      const float* wr = wig + (size_t)e * L2Q * L2Q;
      for (int j = 0; j < L2Q; ++j){
        float wj = wr[j];
        uint4 a0 = *(const uint4*)(hs + j * CC);
        uint4 a1 = *(const uint4*)(hs + j * CC + 8);
        uint4 b0 = *(const uint4*)(hd + j * CC);
        uint4 b1 = *(const uint4*)(hd + j * CC + 8);
        unsigned au[8] = {a0.x,a0.y,a0.z,a0.w,a1.x,a1.y,a1.z,a1.w};
        unsigned bu[8] = {b0.x,b0.y,b0.z,b0.w,b1.x,b1.y,b1.z,b1.w};
#pragma unroll
        for (int i = 0; i < 8; ++i){
          m0[2*i]   += wj * (bf2f(au[i] & 0xffffu) + bf2f(bu[i] & 0xffffu));
          m0[2*i+1] += wj * (bf2f(au[i] >> 16)     + bf2f(bu[i] >> 16));
        }
      }
    }
#pragma unroll
    for (int i = 0; i < 16; ++i){
      int c = part*16 + i;
      efA[row*CC + c] = f2bf(m0[i] * bf2f((unsigned)radS[row*CC + c]));
    }
  }
  __syncthreads();
  for (int ch = 0; ch < 4; ++ch){ // a = silu(xa @ Wa1) . wa2 -> logits
    int nt = ch * 8 + w;
    v8f c0 = {}, c1 = {};
#pragma unroll
    for (int kt = 0; kt < 4; ++kt){
      v16bf b  = frag_b(Wa1T, nt*16, CC, kt*32, lane);
      v16bf a0 = frag_a(efA, 0,  CC, kt*32, lane);
      v16bf a1 = frag_a(efA, 16, CC, kt*32, lane);
      c0 = wmma_bf16(a0, b, c0); c1 = wmma_bf16(a1, b, c1);
    }
    int col = nt*16 + (lane & 15);
    int head = col >> 6;
    float wv = wa2[head * AA + (col & 63)];
    int mb = (lane & 16) >> 1;
#pragma unroll
    for (int v = 0; v < 8; ++v){
      atomicAdd(&lgS[(mb + v)      * HH + head], siluf(c0[v]) * wv);
      atomicAdd(&lgS[(16 + mb + v) * HH + head], siluf(c1[v]) * wv);
    }
  }
  __syncthreads();
  {
    int e = e0 + (t >> 3); int h = t & 7;
    if (e < EE){
      float lg = lgS[t];
      logits[(size_t)e * HH + h] = lg;
      atomicMaxF(&nmax[ei[EE + e] * HH + h], lg);
    }
  }
}

__global__ void k_p(const int* __restrict__ ei, const float* __restrict__ logits,
                    const float* __restrict__ nmax, float* __restrict__ pbuf,
                    float* __restrict__ nz){
  int idx = blockIdx.x * 256 + threadIdx.x;
  if (idx >= EE * HH) return;
  int e = idx >> 3, h = idx & 7;
  int dn = ei[EE + e];
  float p = __expf(logits[idx] - nmax[dn * HH + h]);
  pbuf[idx] = p;
  atomicAdd(&nz[dn * HH + h], p);
}

// ---------------- heavy message kernel: rotate -> Wval -> alpha -> Wout -> rotate back -> scatter ----------------
__global__ void k_attn(const int* __restrict__ ei, const float* __restrict__ wig,
                       const u16* __restrict__ hbuf, const u16* __restrict__ radb,
                       const float* __restrict__ pbuf, const float* __restrict__ nz,
                       const u16* __restrict__ WvalT, const u16* __restrict__ WoutT,
                       float* __restrict__ xout){
  extern __shared__ char smem[];
  u16* sWv  = (u16*)smem;          // 128*128
  u16* sWo  = sWv + 16384;         // 128*128
  u16* wA   = sWo + 16384;         // 32*32
  u16* wAT  = wA + 1024;           // 32*32
  u16* msgT = wAT + 1024;          // [128 n][32 k]
  u16* rot  = msgT + 4096;         // [32 m][128 k]
  u16* vbuf = rot + 4096;          // [32 m][128 k]
  u16* oT   = vbuf + 4096;         // [128 n][32 k]
  float* sAl = (float*)(oT + 4096);
  int t = threadIdx.x, lane = t & 31, w = t >> 5;
  for (int idx = t; idx < 8192; idx += 256){
    ((unsigned*)sWv)[idx] = ((const unsigned*)WvalT)[idx];
    ((unsigned*)sWo)[idx] = ((const unsigned*)WoutT)[idx];
  }
  for (int idx = t; idx < 4096; idx += 256) msgT[idx] = 0; // K-pads stay zero
  __syncthreads();
#pragma unroll 1
  for (int i = 0; i < EPB; ++i){
    int e = blockIdx.x * EPB + i;
    int s = ei[e], dn = ei[EE + e];
    const float* wr = wig + (size_t)e * L2Q * L2Q;
    for (int idx = t; idx < 1024; idx += 256){
      int r = idx >> 5, c = idx & 31;
      u16 v = (r < L2Q && c < L2Q) ? f2bf(wr[r * L2Q + c]) : (u16)0;
      wA[idx] = v;
      wAT[(c << 5) | r] = v;
    }
    { // msg gather, 2 channels per 32-bit load
      const unsigned* hs = (const unsigned*)(hbuf + (size_t)s  * L2Q * CC);
      const unsigned* hd = (const unsigned*)(hbuf + (size_t)dn * L2Q * CC);
      for (int idx = t; idx < (L2Q * CC) / 2; idx += 256){
        int j = idx / 64;            // row
        int cp = idx & 63;           // channel pair
        unsigned a = hs[idx], b = hd[idx];
        int c0i = cp * 2;
        msgT[((c0i    ) << 5) | j] = f2bf(bf2f(a & 0xffffu) + bf2f(b & 0xffffu));
        msgT[((c0i + 1) << 5) | j] = f2bf(bf2f(a >> 16)     + bf2f(b >> 16));
      }
    }
    if (t < HH) sAl[t] = pbuf[(size_t)e * HH + t] / (nz[dn * HH + t] + 1e-9f);
    if (t == 0 && e + 1 < EE) __builtin_prefetch(wig + (size_t)(e + 1) * L2Q * L2Q, 0, 1);
    __syncthreads();
    float radv = bf2f((unsigned)radb[(size_t)e * CC + w*16 + (lane & 15)]);
    { // rot = wig @ msg, * rad
      v8f c0 = {}, c1 = {};
      v16bf b  = frag_b(msgT, w*16, 32, 0, lane);
      v16bf a0 = frag_a(wA, 0,  32, 0, lane);
      v16bf a1 = frag_a(wA, 16, 32, 0, lane);
      c0 = wmma_bf16(a0, b, c0); c1 = wmma_bf16(a1, b, c1);
#pragma unroll
      for (int v = 0; v < 8; ++v){ c0[v] *= radv; c1[v] *= radv; }
      store_acc(rot, 0,  w*16, CC, c0, lane);
      store_acc(rot, 16, w*16, CC, c1, lane);
    }
    __syncthreads();
    { // v = rot @ Wval, * alpha[head]; head == wave (16 cols per head)
      float al = sAl[w];
      v8f c0 = {}, c1 = {};
#pragma unroll
      for (int kt = 0; kt < 4; ++kt){
        v16bf b  = frag_b(sWv, w*16, CC, kt*32, lane);
        v16bf a0 = frag_a(rot, 0,  CC, kt*32, lane);
        v16bf a1 = frag_a(rot, 16, CC, kt*32, lane);
        c0 = wmma_bf16(a0, b, c0); c1 = wmma_bf16(a1, b, c1);
      }
#pragma unroll
      for (int v = 0; v < 8; ++v){ c0[v] *= al; c1[v] *= al; }
      store_acc(vbuf, 0,  w*16, CC, c0, lane);
      store_acc(vbuf, 16, w*16, CC, c1, lane);
    }
    __syncthreads();
    { // o = v @ Wout -> transposed for next B
      v8f c0 = {}, c1 = {};
#pragma unroll
      for (int kt = 0; kt < 4; ++kt){
        v16bf b  = frag_b(sWo, w*16, CC, kt*32, lane);
        v16bf a0 = frag_a(vbuf, 0,  CC, kt*32, lane);
        v16bf a1 = frag_a(vbuf, 16, CC, kt*32, lane);
        c0 = wmma_bf16(a0, b, c0); c1 = wmma_bf16(a1, b, c1);
      }
      store_accT(oT, 0,  w*16, 32, c0, lane);
      store_accT(oT, 16, w*16, 32, c1, lane);
    }
    __syncthreads();
    { // out = wig^T @ o ; atomic scatter into x[dst]
      v8f c0 = {}, c1 = {};
      v16bf b  = frag_b(oT, w*16, 32, 0, lane);
      v16bf a0 = frag_a(wAT, 0,  32, 0, lane);
      v16bf a1 = frag_a(wAT, 16, 32, 0, lane);
      c0 = wmma_bf16(a0, b, c0); c1 = wmma_bf16(a1, b, c1);
      float* xb = xout + (size_t)dn * L2Q * CC;
      int n = w*16 + (lane & 15);
      int mb = (lane & 16) >> 1;
#pragma unroll
      for (int v = 0; v < 8; ++v){
        int m  = mb + v;      if (m  < L2Q) atomicAdd(xb + m  * CC + n, c0[v]);
        int m2 = 16 + mb + v; if (m2 < L2Q) atomicAdd(xb + m2 * CC + n, c1[v]);
      }
    }
    __syncthreads();
  }
}

// ---------------- gated FFN per node ----------------
__global__ void k_ffn(const float* __restrict__ gamma, const u16* __restrict__ Wf1T,
                      const float* __restrict__ Wg, const u16* __restrict__ Wf2T,
                      float* __restrict__ xout){
  extern __shared__ char smem[];
  u16* sW1 = (u16*)smem;         // 128*128
  u16* sW2 = sW1 + 16384;        // 128*128
  u16* hA  = sW2 + 16384;        // 32*128
  u16* hid = hA + 4096;          // 32*128
  float* rms  = (float*)(hid + 4096); // 128
  float* h0   = rms + 128;            // 128
  float* gate = h0 + 128;             // 128
  int t = threadIdx.x, lane = t & 31, w = t >> 5;
  for (int idx = t; idx < 8192; idx += 256){
    ((unsigned*)sW1)[idx] = ((const unsigned*)Wf1T)[idx];
    ((unsigned*)sW2)[idx] = ((const unsigned*)Wf2T)[idx];
  }
  __syncthreads();
#pragma unroll 1
  for (int i = 0; i < NPB; ++i){
    int node = blockIdx.x * NPB + i;
    float* xb = xout + (size_t)node * L2Q * CC;
    if (t < CC){
      float s = 0.0f;
      for (int l = 0; l < L2Q; ++l){ float v = xb[l * CC + t]; s += v * v; }
      float rinv = rsqrtf(s * (1.0f / L2Q) + 1e-6f);
      rms[t] = rinv;
      h0[t] = xb[t] * rinv * gamma[t];
    }
    __syncthreads();
    if (t < FFD){
      float g = 0.0f;
      for (int c = 0; c < CC; ++c) g += h0[c] * Wg[(size_t)c * FFD + t];
      gate[t] = siluf(g);
    }
    for (int idx = t; idx < 32 * CC; idx += 256){
      int l = idx >> 7, c = idx & 127;
      hA[idx] = (l < L2Q) ? f2bf(xb[l * CC + c] * rms[c] * gamma[c]) : (u16)0;
    }
    __syncthreads();
    {
      v8f c0 = {}, c1 = {};
#pragma unroll
      for (int kt = 0; kt < 4; ++kt){
        v16bf b  = frag_b(sW1, w*16, CC, kt*32, lane);
        v16bf a0 = frag_a(hA, 0,  CC, kt*32, lane);
        v16bf a1 = frag_a(hA, 16, CC, kt*32, lane);
        c0 = wmma_bf16(a0, b, c0); c1 = wmma_bf16(a1, b, c1);
      }
      float gv = gate[w*16 + (lane & 15)];
#pragma unroll
      for (int v = 0; v < 8; ++v){ c0[v] *= gv; c1[v] *= gv; }
      store_acc(hid, 0,  w*16, CC, c0, lane);
      store_acc(hid, 16, w*16, CC, c1, lane);
    }
    __syncthreads();
    {
      v8f c0 = {}, c1 = {};
#pragma unroll
      for (int kt = 0; kt < 4; ++kt){
        v16bf b  = frag_b(sW2, w*16, CC, kt*32, lane);
        v16bf a0 = frag_a(hid, 0,  CC, kt*32, lane);
        v16bf a1 = frag_a(hid, 16, CC, kt*32, lane);
        c0 = wmma_bf16(a0, b, c0); c1 = wmma_bf16(a1, b, c1);
      }
      int n = w*16 + (lane & 15); int mb = (lane & 16) >> 1;
#pragma unroll
      for (int v = 0; v < 8; ++v){
        int m  = mb + v;      if (m  < L2Q) xb[m  * CC + n] += c0[v];
        int m2 = 16 + mb + v; if (m2 < L2Q) xb[m2 * CC + n] += c1[v];
      }
    }
    __syncthreads();
  }
}

extern "C" void kernel_launch(void* const* d_in, const int* in_sizes, int n_in,
                              void* d_out, int out_size, void* d_ws, size_t ws_size,
                              hipStream_t stream){
  const int*   zn   = (const int*)d_in[0];
  const float* pos  = (const float*)d_in[1];
  const int*   ei   = (const int*)d_in[2];
  const float* wig  = (const float*)d_in[3];
  const float* aemb = (const float*)d_in[4];
  const float* semb = (const float*)d_in[5];
  const float* demb = (const float*)d_in[6];
  const float* We1  = (const float*)d_in[7];
  const float* We2  = (const float*)d_in[8];
  const float* gA   = (const float*)d_in[9];
  const float* gF   = (const float*)d_in[10];
  const float* Wrad = (const float*)d_in[11];
  const float* Wa1  = (const float*)d_in[12];
  const float* wa2  = (const float*)d_in[13];
  const float* Wval = (const float*)d_in[14];
  const float* Wout = (const float*)d_in[15];
  const float* Wf1  = (const float*)d_in[16];
  const float* Wg   = (const float*)d_in[17];
  const float* Wf2  = (const float*)d_in[18];
  float* x = (float*)d_out;

  char* ws = (char*)d_ws;
  size_t off = 0;
  auto alloc = [&](size_t bytes)->char*{
    char* p = ws + off; off = (off + bytes + 255) & ~(size_t)255; return p;
  };
  u16* efb     = (u16*)alloc((size_t)EE * CC * 2);
  u16* hbuf    = (u16*)alloc((size_t)NN * L2Q * CC * 2);
  u16* radb    = (u16*)alloc((size_t)EE * CC * 2);
  float* logit = (float*)alloc((size_t)EE * HH * 4);
  float* pbuf  = (float*)alloc((size_t)EE * HH * 4);
  float* nmax  = (float*)alloc((size_t)NN * HH * 4);
  float* nz    = (float*)alloc((size_t)NN * HH * 4);
  u16* We1T    = (u16*)alloc((size_t)CC * KE * 2);
  u16* We2T    = (u16*)alloc((size_t)CC * CC * 2);
  u16 *WradT[NLAY], *Wa1T[NLAY], *WvalT[NLAY], *WoutT[NLAY], *Wf1T[NLAY], *Wf2T[NLAY];
  for (int l = 0; l < NLAY; ++l){
    WradT[l] = (u16*)alloc((size_t)CC * CC * 2);
    Wa1T[l]  = (u16*)alloc((size_t)512 * CC * 2);
    WvalT[l] = (u16*)alloc((size_t)CC * CC * 2);
    WoutT[l] = (u16*)alloc((size_t)CC * CC * 2);
    Wf1T[l]  = (u16*)alloc((size_t)CC * CC * 2);
    Wf2T[l]  = (u16*)alloc((size_t)CC * CC * 2);
  }
  auto cvt = [&](const float* W, u16* WT, int K, int Nn, int Kpad){
    int total = Nn * Kpad;
    k_w2bf_t<<<dim3((total + 255) / 256), dim3(256), 0, stream>>>(W, WT, K, Nn, Kpad, total);
  };
  cvt(We1, We1T, 856, 128, KE);
  cvt(We2, We2T, 128, 128, 128);
  for (int l = 0; l < NLAY; ++l){
    cvt(Wrad + (size_t)l*CC*CC,  WradT[l], 128, 128, 128);
    cvt(Wa1  + (size_t)l*CC*512, Wa1T[l],  128, 512, 128);
    cvt(Wval + (size_t)l*CC*CC,  WvalT[l], 128, 128, 128);
    cvt(Wout + (size_t)l*CC*CC,  WoutT[l], 128, 128, 128);
    cvt(Wf1  + (size_t)l*CC*CC,  Wf1T[l],  128, 128, 128);
    cvt(Wf2  + (size_t)l*CC*CC,  Wf2T[l],  128, 128, 128);
  }
  {
    size_t sh = (size_t)32*KE*2 + (size_t)32*CC*2 + 32*4 + 32*4*2;
    k_edgefeat<<<dim3((EE + 31) / 32), dim3(256), sh, stream>>>(ei, zn, pos, semb, demb,
                                                               We1T, We2T, efb);
  }
  k_embed<<<dim3((NN*L2Q*CC + 255) / 256), dim3(256), 0, stream>>>(zn, aemb, x);
  for (int l = 0; l < NLAY; ++l){
    k_init_soft<<<dim3((NN*HH + 255) / 256), dim3(256), 0, stream>>>(nmax, nz);
    k_rmsnorm<<<dim3((NN*CC + 255) / 256), dim3(256), 0, stream>>>(x, gA + (size_t)l*CC, hbuf);
    {
      size_t sh = (size_t)32*CC*2*2 + 256*4;
      k_logits<<<dim3((EE + 31) / 32), dim3(256), sh, stream>>>(ei, wig, hbuf, efb,
          WradT[l], Wa1T[l], wa2 + (size_t)l*HH*AA, radb, logit, nmax);
    }
    k_p<<<dim3((EE*HH + 255) / 256), dim3(256), 0, stream>>>(ei, logit, nmax, pbuf, nz);
    {
      size_t sh = (size_t)(16384*2 + 1024*2 + 4096*4) * 2 + 8*4;
      k_attn<<<dim3(EE / EPB), dim3(256), sh, stream>>>(ei, wig, hbuf, radb, pbuf, nz,
                                                        WvalT[l], WoutT[l], x);
    }
    {
      size_t sh = (size_t)(16384*2 + 4096*2) * 2 + 128*4*3;
      k_ffn<<<dim3(NN / NPB), dim3(256), sh, stream>>>(gF + (size_t)l*CC, Wf1T[l],
                                                       Wg + (size_t)l*CC*FFD, Wf2T[l], x);
    }
  }
  (void)in_sizes; (void)n_in; (void)out_size; (void)ws_size;
}
